// MAB_30760555774319
// MI455X (gfx1250) — matrix operand order repaired and checked by reading, not compiled, not run on hardware
//
#include <hip/hip_runtime.h>
#include <hip/hip_bf16.h>

typedef __attribute__((ext_vector_type(16))) _Float16 v16h;
typedef __attribute__((ext_vector_type(8)))  _Float16 v8h;
typedef __attribute__((ext_vector_type(8)))  float    v8f;

#define NTOK   2048
#define DMODEL 512
#define NHEAD  8
#define DHEAD  64
#define DFFN   2048
#define BATCH  4
#define ROWS   (BATCH * NTOK)                 // 8192
#define ATTN_SCALE 0.044194173824159216f      // 1/sqrt(512)

// ---------------------------------------------------------------------------
// Fragment loaders (wave32 WMMA; k order below is this file's contiguous one, same in A and B, not ISA table order)
// A 16x32 f16: lane L holds row L%16; lo lanes K 0..15, hi lanes
// K 16..31  -> two 16B chunks at p (incl. hi*16) and p+8.
// B 32x16 f16: lane L holds col L%16; lo lanes K 0..15, hi lanes K 16..31
// -> one contiguous 32B chunk at p (incl. hi*16).
// C/D 16x16 f32: vgpr r, lo lanes row r, hi lanes row r+8; col = lane%16.
// ---------------------------------------------------------------------------
__device__ __forceinline__ v16h ldA16(const _Float16* p) {
  v8h x = *(const v8h*)p;
  v8h y = *(const v8h*)(p + 8);
  return __builtin_shufflevector(x, y, 0,1,2,3,4,5,6,7,8,9,10,11,12,13,14,15);
}
__device__ __forceinline__ v16h ldB16(const _Float16* p) {
  return *(const v16h*)p;
}
__device__ __forceinline__ v8f wmma_f16(v16h a, v16h b, v8f c) {
  return __builtin_amdgcn_wmma_f32_16x16x32_f16(false, a, false, b, (short)0, c,
                                                false, false);
}
__device__ __forceinline__ float rmax16(float v) {
#pragma unroll
  for (int o = 1; o < 16; o <<= 1) v = fmaxf(v, __shfl_xor(v, o, 32));
  return v;
}
__device__ __forceinline__ float rsum16(float v) {
#pragma unroll
  for (int o = 1; o < 16; o <<= 1) v += __shfl_xor(v, o, 32);
  return v;
}

// ---------------------------------------------------------------------------
// Generic WMMA GEMM: C[M x Nout] = A16[M x K] * Bt16[Nout x K]^T (+bias)
// Each wave: one 32x64 strip -> 2 A frags + 4 B frags feed 8 WMMAs per
// 32-deep k step (8 independent accumulator chains, B reused across rows).
// Epilogue: optional fp32 residual, exact GELU, fp32 / f16 / f16-transposed out.
// ---------------------------------------------------------------------------
__global__ __launch_bounds__(256) void wmma_gemm_kernel(
    const _Float16* __restrict__ A, const _Float16* __restrict__ Bt,
    const float* __restrict__ bias, const float* __restrict__ res,
    float* __restrict__ outF, _Float16* __restrict__ outH,
    _Float16* __restrict__ outHT, int M, int Nout, int K, int ldT, int gelu) {
  const int wid = threadIdx.x >> 5;
  const int lane = threadIdx.x & 31;
  const int l = lane & 15, hi = lane >> 4;
  const int tilesN4 = Nout >> 6;
  const int id = blockIdx.x * 8 + wid;
  const int tm = id / tilesN4;
  const int tn = id % tilesN4;
  const int row0 = tm << 5;   // 32-row strip
  const int col0 = tn << 6;   // 64-col strip
  if (row0 >= M) return;  // wave-uniform

  v8f acc[8] = {};
  const _Float16* ap0 = A + (size_t)(row0 + l) * K + hi * 16;
  const _Float16* ap1 = ap0 + (size_t)16 * K;
  const _Float16* bp = Bt + (size_t)(col0 + l) * K + hi * 16;
  const size_t bstep = (size_t)16 * K;
  for (int k0 = 0; k0 < K; k0 += 32) {
    const v16h a0 = ldA16(ap0 + k0);
    const v16h a1 = ldA16(ap1 + k0);
#pragma unroll
    for (int t = 0; t < 4; ++t) {
      const v16h b = ldB16(bp + (size_t)t * bstep + k0);
      acc[t] = wmma_f16(a0, b, acc[t]);
      acc[4 + t] = wmma_f16(a1, b, acc[4 + t]);
    }
  }
#pragma unroll
  for (int m = 0; m < 2; ++m) {
#pragma unroll
    for (int t = 0; t < 4; ++t) {
      const int col = col0 + t * 16 + l;
      const float bv = bias ? bias[col] : 0.0f;
      const v8f cv = acc[m * 4 + t];
#pragma unroll
      for (int r = 0; r < 8; ++r) {
        const int row = row0 + m * 16 + r + hi * 8;
        float v = cv[r] + bv;
        const size_t idx = (size_t)row * Nout + col;
        if (res) v += res[idx];
        if (gelu) v = 0.5f * v * (1.0f + erff(v * 0.70710678118654752f));
        if (outF) outF[idx] = v;
        if (outH) outH[idx] = (_Float16)v;
        if (outHT) outHT[(size_t)col * ldT + row] = (_Float16)v;
      }
    }
  }
}

// ---------------------------------------------------------------------------
// Flash-style attention. One wave handles 16 query rows of one (b,h).
// Q16:[B*N][D] f16, K16:[B*N][D] f16, Vt16:[B*H][DHEAD][N] f16 (transposed),
// O16:[B*N][D] f16.
// ---------------------------------------------------------------------------
__global__ __launch_bounds__(256) void attn_kernel(
    const _Float16* __restrict__ Q16, const _Float16* __restrict__ K16,
    const _Float16* __restrict__ Vt16, _Float16* __restrict__ O16) {
  const int wid = threadIdx.x >> 5;
  const int lane = threadIdx.x & 31;
  const int l = lane & 15, hi = lane >> 4;
  const int id = blockIdx.x * 8 + wid;  // B*H*(N/16) = 4096 wave tiles
  const int qt = id & 127;              // N/16
  const int bh = id >> 7;               // 0..31
  const int b = bh >> 3, h = bh & 7;
  const int q0 = qt << 4;

  __shared__ __align__(16) _Float16 Pbuf[8][16 * 32];
  _Float16* pp = &Pbuf[wid][0];

  // Q A-fragments for the two 32-wide d steps (d = 0..63), kept in registers.
  const _Float16* qb =
      Q16 + (size_t)(b * NTOK + q0 + l) * DMODEL + h * DHEAD + hi * 16;
  const v16h qa0 = ldA16(qb);
  const v16h qa1 = ldA16(qb + 32);

  v8f o0 = {}, o1 = {}, o2 = {}, o3 = {};
  float mrow[8], lrow[8];
#pragma unroll
  for (int r = 0; r < 8; ++r) { mrow[r] = -3.0e38f; lrow[r] = 0.0f; }

  const _Float16* kb =
      K16 + (size_t)(b * NTOK + l) * DMODEL + h * DHEAD + hi * 16;
  const _Float16* vb = Vt16 + (size_t)(bh * DHEAD + l) * NTOK + hi * 16;

  for (int j = 0; j < NTOK; j += 32) {
    // S = Q K^T for key columns j..j+31 (two 16x16 tiles, K-dim = d = 64)
    const _Float16* k0p = kb + (size_t)j * DMODEL;
    const _Float16* k1p = k0p + (size_t)16 * DMODEL;
    v8f s0 = {}, s1 = {};
    s0 = wmma_f16(qa0, ldB16(k0p), s0);
    s0 = wmma_f16(qa1, ldB16(k0p + 32), s0);
    s1 = wmma_f16(qa0, ldB16(k1p), s1);
    s1 = wmma_f16(qa1, ldB16(k1p + 32), s1);

    // online softmax; each lane: 1 key column x 8 rows; row stats via 16-lane
    // shuffles (rows of S match rows of O accumulator layout).
#pragma unroll
    for (int r = 0; r < 8; ++r) {
      const float a = s0[r] * ATTN_SCALE;
      const float c = s1[r] * ATTN_SCALE;
      const float mnew = fmaxf(mrow[r], rmax16(fmaxf(a, c)));
      const float alpha = __expf(mrow[r] - mnew);
      const float e0 = __expf(a - mnew);
      const float e1 = __expf(c - mnew);
      lrow[r] = lrow[r] * alpha + rsum16(e0 + e1);
      mrow[r] = mnew;
      o0[r] *= alpha; o1[r] *= alpha; o2[r] *= alpha; o3[r] *= alpha;
      const int row = r + hi * 8;
      pp[row * 32 + l] = (_Float16)e0;        // C-layout -> LDS row-major
      pp[row * 32 + 16 + l] = (_Float16)e1;
    }
    // reload P as an A fragment (per-wave LDS region; DS ops in-order per wave)
    const v16h pa = ldA16(pp + l * 32 + hi * 16);
    const _Float16* vp = vb + j;
    o0 = wmma_f16(pa, ldB16(vp), o0);
    o1 = wmma_f16(pa, ldB16(vp + (size_t)16 * NTOK), o1);
    o2 = wmma_f16(pa, ldB16(vp + (size_t)32 * NTOK), o2);
    o3 = wmma_f16(pa, ldB16(vp + (size_t)48 * NTOK), o3);
  }

#pragma unroll
  for (int r = 0; r < 8; ++r) {
    const float inv = 1.0f / lrow[r];
    const int row = q0 + r + hi * 8;
    const size_t base = (size_t)(b * NTOK + row) * DMODEL + h * DHEAD + l;
    O16[base] = (_Float16)(o0[r] * inv);
    O16[base + 16] = (_Float16)(o1[r] * inv);
    O16[base + 32] = (_Float16)(o2[r] * inv);
    O16[base + 48] = (_Float16)(o3[r] * inv);
  }
}

// ---------------------------------------------------------------------------
// LayerNorm over last dim (512), fp32 in -> f16 out. One block per row.
// ---------------------------------------------------------------------------
__global__ __launch_bounds__(256) void ln_kernel(const float* __restrict__ X,
                                                 const float* __restrict__ g,
                                                 const float* __restrict__ be,
                                                 _Float16* __restrict__ out) {
  const int row = blockIdx.x;
  const int t = threadIdx.x;
  const float* xr = X + (size_t)row * DMODEL;
  const float a = xr[t], c = xr[t + 256];
  __shared__ float ssum[256], ssq[256];
  ssum[t] = a + c;
  ssq[t] = a * a + c * c;
  __syncthreads();
  for (int s = 128; s > 0; s >>= 1) {
    if (t < s) { ssum[t] += ssum[t + s]; ssq[t] += ssq[t + s]; }
    __syncthreads();
  }
  const float mean = ssum[0] * (1.0f / DMODEL);
  const float var = ssq[0] * (1.0f / DMODEL) - mean * mean;
  const float inv = rsqrtf(var + 1e-5f);
  _Float16* orow = out + (size_t)row * DMODEL;
  orow[t] = (_Float16)((a - mean) * inv * g[t] + be[t]);
  orow[t + 256] = (_Float16)((c - mean) * inv * g[t + 256] + be[t + 256]);
}

__global__ void f32_to_f16_kernel(const float* __restrict__ in,
                                  _Float16* __restrict__ out, int n) {
  const int i = blockIdx.x * blockDim.x + threadIdx.x;
  if (i < n) out[i] = (_Float16)in[i];
}

// W[K][Nout] fp32 -> Wt[Nout][K] f16
__global__ void transpose_f16_kernel(const float* __restrict__ in,
                                     _Float16* __restrict__ out, int K,
                                     int Nout) {
  const int i = blockIdx.x * blockDim.x + threadIdx.x;
  if (i < K * Nout) {
    const int k = i / Nout, n = i % Nout;
    out[(size_t)n * K + k] = (_Float16)in[i];
  }
}

// ---------------------------------------------------------------------------
extern "C" void kernel_launch(void* const* d_in, const int* in_sizes, int n_in,
                              void* d_out, int out_size, void* d_ws,
                              size_t ws_size, hipStream_t stream) {
  (void)in_sizes; (void)n_in; (void)out_size; (void)ws_size;
  const float* X = (const float*)d_in[0];
  const float* Y = (const float*)d_in[1];
  const float* Wq = (const float*)d_in[2];
  const float* bq = (const float*)d_in[3];
  const float* Wk = (const float*)d_in[4];
  const float* bk = (const float*)d_in[5];
  const float* Wv = (const float*)d_in[6];
  const float* bv = (const float*)d_in[7];
  const float* Wm = (const float*)d_in[8];
  const float* bm = (const float*)d_in[9];
  const float* g0 = (const float*)d_in[10];
  const float* b0 = (const float*)d_in[11];
  const float* g1 = (const float*)d_in[12];
  const float* b1 = (const float*)d_in[13];
  const float* W1 = (const float*)d_in[14];
  const float* bb1 = (const float*)d_in[15];
  const float* W2 = (const float*)d_in[16];
  const float* bb2 = (const float*)d_in[17];
  float* out = (float*)d_out;

  char* ws = (char*)d_ws;
  size_t off = 0;
  auto take = [&](size_t bytes) -> void* {
    void* p = ws + off;
    off += (bytes + 255) & ~(size_t)255;
    return p;
  };
  const size_t act16 = (size_t)ROWS * DMODEL * sizeof(_Float16);
  const size_t act32 = (size_t)ROWS * DMODEL * sizeof(float);
  _Float16* Xn16 = (_Float16*)take(act16);  // LN(X); reused later for LN(Hx)
  _Float16* Y16 = (_Float16*)take(act16);   // Y f16; reused for attn output
  _Float16* Q16 = (_Float16*)take(act16);
  float* Q32 = (float*)take(act32);         // residual
  _Float16* K16 = (_Float16*)take(act16);
  _Float16* Vt16 = (_Float16*)take(act16);  // [B*H][DHEAD][N]
  float* Hx32 = (float*)take(act32);
  _Float16* FF1 = (_Float16*)take((size_t)ROWS * DFFN * sizeof(_Float16));
  _Float16* Wq16 = (_Float16*)take((size_t)DMODEL * DMODEL * 2);
  _Float16* Wk16 = (_Float16*)take((size_t)DMODEL * DMODEL * 2);
  _Float16* Wv16 = (_Float16*)take((size_t)DMODEL * DMODEL * 2);
  _Float16* Wm16 = (_Float16*)take((size_t)DMODEL * DMODEL * 2);
  _Float16* W1t = (_Float16*)take((size_t)DMODEL * DFFN * 2);
  _Float16* W2t = (_Float16*)take((size_t)DFFN * DMODEL * 2);

  auto gemm = [&](const _Float16* A, const _Float16* Bt, const float* bias,
                  const float* res, float* oF, _Float16* oH, _Float16* oHT,
                  int M, int Nout, int K, int ldT, int gelu) {
    const int blocks = (M / 32) * (Nout / 64) / 8;
    wmma_gemm_kernel<<<blocks, 256, 0, stream>>>(A, Bt, bias, res, oF, oH, oHT,
                                                 M, Nout, K, ldT, gelu);
  };

  // --- prep: weights -> f16 transposed, Y -> f16, LN(X) -> f16
  const int sq = DMODEL * DMODEL;
  transpose_f16_kernel<<<(sq + 255) / 256, 256, 0, stream>>>(Wq, Wq16, DMODEL, DMODEL);
  transpose_f16_kernel<<<(sq + 255) / 256, 256, 0, stream>>>(Wk, Wk16, DMODEL, DMODEL);
  transpose_f16_kernel<<<(sq + 255) / 256, 256, 0, stream>>>(Wv, Wv16, DMODEL, DMODEL);
  transpose_f16_kernel<<<(sq + 255) / 256, 256, 0, stream>>>(Wm, Wm16, DMODEL, DMODEL);
  transpose_f16_kernel<<<(DMODEL * DFFN + 255) / 256, 256, 0, stream>>>(W1, W1t, DMODEL, DFFN);
  transpose_f16_kernel<<<(DFFN * DMODEL + 255) / 256, 256, 0, stream>>>(W2, W2t, DFFN, DMODEL);
  f32_to_f16_kernel<<<(ROWS * DMODEL + 255) / 256, 256, 0, stream>>>(Y, Y16, ROWS * DMODEL);
  ln_kernel<<<ROWS, 256, 0, stream>>>(X, g0, b0, Xn16);

  // --- projections
  gemm(Xn16, Wq16, bq, nullptr, Q32, Q16, nullptr, ROWS, DMODEL, DMODEL, 0, 0);
  gemm(Y16, Wk16, bk, nullptr, nullptr, K16, nullptr, ROWS, DMODEL, DMODEL, 0, 0);
  for (int b = 0; b < BATCH; ++b)  // per-batch so transposed V layout is flat
    gemm(Y16 + (size_t)b * NTOK * DMODEL, Wv16, bv, nullptr, nullptr, nullptr,
         Vt16 + (size_t)b * DMODEL * NTOK, NTOK, DMODEL, DMODEL, NTOK, 0);

  // --- attention (writes f16 attn output into Y16, which is now free)
  attn_kernel<<<(BATCH * NHEAD * (NTOK / 16)) / 8, 256, 0, stream>>>(Q16, K16, Vt16, Y16);

  // --- mix heads + residual(Q), LN, FFN
  gemm(Y16, Wm16, bm, Q32, Hx32, nullptr, nullptr, ROWS, DMODEL, DMODEL, 0, 0);
  ln_kernel<<<ROWS, 256, 0, stream>>>(Hx32, g1, b1, Xn16);
  gemm(Xn16, W1t, bb1, nullptr, nullptr, FF1, nullptr, ROWS, DFFN, DMODEL, 0, 1);
  gemm(FF1, W2t, bb2, Hx32, out, nullptr, nullptr, ROWS, DMODEL, DFFN, 0, 0);
}
